// ProbAttention_4458176053855
// MI455X (gfx1250) — matrix operand order repaired and checked by reading, not compiled, not run on hardware
//
#include <hip/hip_runtime.h>
#include <hip/hip_bf16.h>

// ---------------- problem constants (from reference) ----------------
#define BB   4
#define LL   2048
#define DD   512
#define SS   2048
#define HH   8
#define EE   64
#define BH   (BB*HH)         // 32
#define SAMPK 40
#define NTOP  40
#define UPAD  48             // NTOP padded to 3 WMMA row-tiles

typedef __attribute__((ext_vector_type(16))) __bf16 v16bf;
typedef __attribute__((ext_vector_type(8)))  float  v8f;
typedef __attribute__((ext_vector_type(4)))  unsigned int u32x4;

union FragBF { u32x4 u[2]; v16bf v; };

__device__ __forceinline__ unsigned short f2bf(float f) {
    unsigned u = __builtin_bit_cast(unsigned, f);
    unsigned r = 0x7FFFu + ((u >> 16) & 1u);   // round-to-nearest-even
    return (unsigned short)((u + r) >> 16);
}

// -------- 1) K -> bf16 row-major ; V -> bf16 transposed (E x S) --------
__global__ void conv_kv_kernel(const float* __restrict__ kmat,
                               const float* __restrict__ vmat,
                               unsigned short* __restrict__ Kbf,
                               unsigned short* __restrict__ VTbf) {
    size_t t = (size_t)blockIdx.x * 256 + threadIdx.x;   // (bh, s, e)
    if (t >= (size_t)BH * SS * EE) return;
    int e  = (int)(t & (EE - 1));
    int s  = (int)((t >> 6) & (SS - 1));
    int bh = (int)(t >> 17);
    int b = bh >> 3, h = bh & 7;
    size_t src = ((size_t)(b * SS + s) * DD) + (size_t)h * EE + e;
    Kbf[t] = f2bf(kmat[src]);
    VTbf[((size_t)bh * EE + e) * SS + s] = f2bf(vmat[src]);
}

// -------- 2a) Vsum[bh][e] = sum_s V[b,s,h*E+e] --------
__global__ void vsum_kernel(const float* __restrict__ vmat,
                            float* __restrict__ Vsum) {
    int bh = blockIdx.x;          // 32 blocks, 64 threads
    int e  = threadIdx.x;
    int b = bh >> 3, h = bh & 7;
    float acc = 0.f;
    for (int s = 0; s < SS; ++s)
        acc += vmat[((size_t)(b * SS + s) * DD) + (size_t)h * EE + e];
    Vsum[bh * EE + e] = acc;
}

// -------- 2b) broadcast fill: out[b,l,h*E+e] = Vsum[bh,e] --------
__global__ void bcast_kernel(const float* __restrict__ Vsum,
                             float* __restrict__ out) {
    size_t t = (size_t)blockIdx.x * 256 + threadIdx.x;   // B*L*D
    if (t >= (size_t)BB * LL * DD) return;
    int d  = (int)(t & (DD - 1));
    int h  = d >> 6, e = d & 63;
    int b  = (int)(t >> 20);                              // t / (L*D)
    out[t] = Vsum[(b * HH + h) * EE + e];
}

// -------- 3) sparsity measure M = max_j(q.k_samp) - sum_j(q.k_samp)/S --------
__global__ void mscore_kernel(const float* __restrict__ q,
                              const float* __restrict__ k,
                              const int*   __restrict__ idx,
                              float* __restrict__ M) {
    int t = blockIdx.x * 256 + threadIdx.x;               // (bh, l)
    if (t >= BH * LL) return;
    int bh = t >> 11, l = t & (LL - 1);
    int b = bh >> 3, h = bh & 7;
    const float4* qv = (const float4*)(q + ((size_t)(b * LL + l) * DD) + (size_t)h * EE);
    float4 qr[16];
#pragma unroll
    for (int i = 0; i < 16; ++i) qr[i] = qv[i];
    float mx = -INFINITY, sm = 0.f;
    for (int j = 0; j < SAMPK; ++j) {
        int s = idx[l * SAMPK + j];
        const float4* kv = (const float4*)(k + ((size_t)(b * SS + s) * DD) + (size_t)h * EE);
        float acc = 0.f;
#pragma unroll
        for (int i = 0; i < 16; ++i) {
            float4 kk = kv[i];
            acc += qr[i].x * kk.x + qr[i].y * kk.y + qr[i].z * kk.z + qr[i].w * kk.w;
        }
        mx = fmaxf(mx, acc);
        sm += acc;
    }
    M[t] = mx - sm * (1.0f / (float)SS);
}

// -------- 4) top-40 per (b,h) + gather Q_reduce -> bf16 (padded to 48 rows) --------
__global__ void topk_gather_kernel(float* __restrict__ M,
                                   int*   __restrict__ Mtop,
                                   const float* __restrict__ q,
                                   unsigned short* __restrict__ Qr) {
    int bh = blockIdx.x;                                  // 32 blocks, 256 threads
    int b = bh >> 3, h = bh & 7;
    __shared__ float sv[256];
    __shared__ int   si[256];
    __shared__ int   stop[UPAD];
    float* Mrow = M + (size_t)bh * LL;
    for (int t = 0; t < NTOP; ++t) {
        float best = -INFINITY; int bi = 0;
        for (int c = threadIdx.x; c < LL; c += 256) {
            float v = Mrow[c];
            if (v > best) { best = v; bi = c; }
        }
        sv[threadIdx.x] = best; si[threadIdx.x] = bi;
        __syncthreads();
        for (int off = 128; off > 0; off >>= 1) {
            if ((int)threadIdx.x < off) {
                float vo = sv[threadIdx.x + off]; int io = si[threadIdx.x + off];
                if (vo > sv[threadIdx.x] || (vo == sv[threadIdx.x] && io < si[threadIdx.x])) {
                    sv[threadIdx.x] = vo; si[threadIdx.x] = io;
                }
            }
            __syncthreads();
        }
        if (threadIdx.x == 0) {
            stop[t] = si[0];
            Mtop[bh * UPAD + t] = si[0];
            Mrow[si[0]] = -INFINITY;                      // exclude for next round
        }
        __syncthreads();
    }
    if (threadIdx.x < UPAD - NTOP) {
        stop[NTOP + threadIdx.x] = 0;
        Mtop[bh * UPAD + NTOP + threadIdx.x] = 0;
    }
    __syncthreads();
    // gather selected Q rows, convert to bf16, pad rows 40..47 with zero
    for (int t = threadIdx.x; t < UPAD * EE; t += 256) {
        int u = t >> 6, e = t & 63;
        float val = 0.f;
        if (u < NTOP) {
            int l = stop[u];
            val = q[((size_t)(b * LL + l) * DD) + (size_t)h * EE + e];
        }
        Qr[((size_t)bh * UPAD + u) * EE + e] = f2bf(val);
    }
}

// -------- 5) scores = scale * Q_r @ K^T  (bf16 WMMA, f32 acc) --------
// grid (BH, 3) x 128 threads (4 waves). Wave handles N-tiles wave,wave+4,...
__global__ void scores_kernel(const unsigned short* __restrict__ Qr,
                              const unsigned short* __restrict__ Kbf,
                              float* __restrict__ scores) {
    int bh = blockIdx.x, ut = blockIdx.y;
    int wave = threadIdx.x >> 5, lane = threadIdx.x & 31;
    int half = lane >> 4, r = lane & 15;

    // A fragments (16x64 Q tile, two k-steps of 32)
    // ISA layout: lane<16 holds row r, K {0..7, 16..23}+ebase; lane>=16: {8..15, 24..31}
    const unsigned short* arow = Qr + ((size_t)(bh * UPAD + ut * 16 + r) * EE);
    FragBF a0, a1;
    a0.u[0] = *(const u32x4*)(arow + 0  + half * 8);
    a0.u[1] = *(const u32x4*)(arow + 16 + half * 8);
    a1.u[0] = *(const u32x4*)(arow + 32 + half * 8);
    a1.u[1] = *(const u32x4*)(arow + 48 + half * 8);

    for (int nt = wave; nt < SS / 16; nt += 4) {
        int s = nt * 16 + r;                               // B column = lane&15
        const unsigned short* krow = Kbf + ((size_t)bh * SS + s) * EE;
        // B layout: lane<16 -> K rows (=e) {0..7, 16..23} of ebase; lane>=16 -> {8..15, 24..31}
        FragBF b0, b1;
        b0.u[0] = *(const u32x4*)(krow + half * 8);
        b0.u[1] = *(const u32x4*)(krow + 16 + half * 8);
        b1.u[0] = *(const u32x4*)(krow + 32 + half * 8);
        b1.u[1] = *(const u32x4*)(krow + 48 + half * 8);
        v8f c = {};
        c = __builtin_amdgcn_wmma_f32_16x16x32_bf16(false, a0.v, false, b0.v,
                                                    (short)0, c, false, false);
        c = __builtin_amdgcn_wmma_f32_16x16x32_bf16(false, a1.v, false, b1.v,
                                                    (short)0, c, false, false);
        float* obase = scores + ((size_t)(bh * UPAD + ut * 16) * SS) + nt * 16 + r;
#pragma unroll
        for (int i = 0; i < 8; ++i) {
            int m = i + half * 8;                          // D row
            obase[(size_t)m * SS] = c[i] * 0.125f;         // 1/sqrt(64)
        }
    }
}

// -------- 6) row softmax: P = exp(s - max) (bf16, unnormalized), denom = sum --------
__global__ void softmax_kernel(const float* __restrict__ scores,
                               unsigned short* __restrict__ Pbf,
                               float* __restrict__ denom) {
    int row = blockIdx.x;                                  // BH*UPAD rows, 256 threads
    const float* srow = scores + (size_t)row * SS;
    __shared__ float red[256];
    float m = -INFINITY;
    for (int c = threadIdx.x; c < SS; c += 256) m = fmaxf(m, srow[c]);
    red[threadIdx.x] = m; __syncthreads();
    for (int off = 128; off > 0; off >>= 1) {
        if ((int)threadIdx.x < off)
            red[threadIdx.x] = fmaxf(red[threadIdx.x], red[threadIdx.x + off]);
        __syncthreads();
    }
    m = red[0]; __syncthreads();
    float s = 0.f;
    unsigned short* prow = Pbf + (size_t)row * SS;
    for (int c = threadIdx.x; c < SS; c += 256) {
        float p = __expf(srow[c] - m);
        s += p;
        prow[c] = f2bf(p);
    }
    red[threadIdx.x] = s; __syncthreads();
    for (int off = 128; off > 0; off >>= 1) {
        if ((int)threadIdx.x < off)
            red[threadIdx.x] += red[threadIdx.x + off];
        __syncthreads();
    }
    if (threadIdx.x == 0) denom[row] = red[0];
}

// -------- 7) ctx = (P @ V) / denom, scattered into out rows Mtop --------
// grid (BH, 3) x 128 threads (4 waves). Wave w computes e-tile w (E/16 = 4).
__global__ void pv_kernel(const unsigned short* __restrict__ Pbf,
                          const unsigned short* __restrict__ VTbf,
                          const float* __restrict__ denom,
                          const int* __restrict__ Mtop,
                          float* __restrict__ out) {
    int bh = blockIdx.x, ut = blockIdx.y;
    int b = bh >> 3, h = bh & 7;
    int wave = threadIdx.x >> 5, lane = threadIdx.x & 31;
    int half = lane >> 4, r = lane & 15;
    int ecol = wave * 16 + r;

    const unsigned short* prow = Pbf  + (size_t)(bh * UPAD + ut * 16 + r) * SS;
    const unsigned short* vrow = VTbf + ((size_t)bh * EE + ecol) * SS;

    v8f c = {};
    for (int ks = 0; ks < SS / 32; ++ks) {
        int s0 = ks * 32;
        FragBF a, bb;
        a.u[0]  = *(const u32x4*)(prow + s0 + half * 8);
        a.u[1]  = *(const u32x4*)(prow + s0 + 16 + half * 8);
        bb.u[0] = *(const u32x4*)(vrow + s0 + half * 8);
        bb.u[1] = *(const u32x4*)(vrow + s0 + 16 + half * 8);
        c = __builtin_amdgcn_wmma_f32_16x16x32_bf16(false, a.v, false, bb.v,
                                                    (short)0, c, false, false);
    }
#pragma unroll
    for (int i = 0; i < 8; ++i) {
        int m = i + half * 8;
        int u = ut * 16 + m;
        if (u < NTOP) {
            int l = Mtop[bh * UPAD + u];
            float d = denom[bh * UPAD + u];
            out[((size_t)(b * LL + l) * DD) + (size_t)h * EE + ecol] = c[i] / d;
        }
    }
}

// ---------------- launcher ----------------
static inline size_t align256(size_t x) { return (x + 255) & ~(size_t)255; }

extern "C" void kernel_launch(void* const* d_in, const int* in_sizes, int n_in,
                              void* d_out, int out_size, void* d_ws, size_t ws_size,
                              hipStream_t stream) {
    (void)in_sizes; (void)n_in; (void)out_size; (void)ws_size;
    const float* q   = (const float*)d_in[0];
    const float* k   = (const float*)d_in[1];
    const float* v   = (const float*)d_in[2];
    const int*   smp = (const int*)d_in[3];
    float* out = (float*)d_out;

    char* ws = (char*)d_ws;
    size_t off = 0;
    unsigned short* Kbf  = (unsigned short*)(ws + off); off = align256(off + (size_t)BH*SS*EE*2);
    unsigned short* VTbf = (unsigned short*)(ws + off); off = align256(off + (size_t)BH*EE*SS*2);
    unsigned short* Qr   = (unsigned short*)(ws + off); off = align256(off + (size_t)BH*UPAD*EE*2);
    float* scores        = (float*)(ws + off);          off = align256(off + (size_t)BH*UPAD*SS*4);
    unsigned short* Pbf  = (unsigned short*)(ws + off); off = align256(off + (size_t)BH*UPAD*SS*2);
    float* M             = (float*)(ws + off);          off = align256(off + (size_t)BH*LL*4);
    int*   Mtop          = (int*)(ws + off);            off = align256(off + (size_t)BH*UPAD*4);
    float* denom         = (float*)(ws + off);          off = align256(off + (size_t)BH*UPAD*4);
    float* Vsum          = (float*)(ws + off);          off = align256(off + (size_t)BH*EE*4);

    // 1) convert K/V to bf16 (V transposed)
    {
        size_t total = (size_t)BH * SS * EE;
        conv_kv_kernel<<<dim3((unsigned)((total + 255) / 256)), dim3(256), 0, stream>>>(k, v, Kbf, VTbf);
    }
    // 2) V row-sum + broadcast fill of output
    vsum_kernel<<<dim3(BH), dim3(EE), 0, stream>>>(v, Vsum);
    {
        size_t total = (size_t)BB * LL * DD;
        bcast_kernel<<<dim3((unsigned)((total + 255) / 256)), dim3(256), 0, stream>>>(Vsum, out);
    }
    // 3) sparsity measure
    mscore_kernel<<<dim3((BH * LL + 255) / 256), dim3(256), 0, stream>>>(q, k, smp, M);
    // 4) top-40 + Q_reduce gather
    topk_gather_kernel<<<dim3(BH), dim3(256), 0, stream>>>(M, Mtop, q, Qr);
    // 5) scores GEMM (WMMA)
    scores_kernel<<<dim3(BH, UPAD / 16), dim3(128), 0, stream>>>(Qr, Kbf, scores);
    // 6) softmax
    softmax_kernel<<<dim3(BH * UPAD), dim3(256), 0, stream>>>(scores, Pbf, denom);
    // 7) P @ V (WMMA) + scatter
    pv_kernel<<<dim3(BH, UPAD / 16), dim3(128), 0, stream>>>(Pbf, VTbf, denom, Mtop, out);
}
